// TransformerBlock_88545045774941
// MI455X (gfx1250) — compile-verified, matrix operand order checked by reading, not run on hardware
//
#include <hip/hip_runtime.h>
#include <hip/hip_bf16.h>

typedef __bf16 bf16;
typedef __attribute__((ext_vector_type(16))) __bf16 v16bf;
typedef __attribute__((ext_vector_type(8)))  __bf16 v8bf;
typedef __attribute__((ext_vector_type(8)))  float   v8f;
typedef __attribute__((ext_vector_type(4)))  float   v4f;
typedef __attribute__((ext_vector_type(4)))  int     v4i;

typedef __attribute__((address_space(1))) v4i* as1_v4i;
typedef __attribute__((address_space(3))) v4i* as3_v4i;

namespace {
constexpr int S_LEN  = 2048;
constexpr int BATCH  = 2;
constexpr int DMODEL = 2048;
constexpr int NHEAD  = 16;
constexpr int DHEAD  = 128;
constexpr int HIDDEN = 5632;
constexpr int MROWS  = S_LEN * BATCH;   // 4096
constexpr float EPSF = 1e-5f;
}

template <bool B> struct BoolC { static constexpr bool value = B; };

// Internal activation row layout is batch-major: m = b*S_LEN + s.
// External (x, d_out) layout is [S, B, D]: row = s*BATCH + b.
__device__ __forceinline__ size_t remap_row(int m) {  // internal m -> external row
  return (size_t)((m & (S_LEN - 1)) * BATCH + (m >> 11));
}

// ---------------------------------------------------------------------------
// CDNA5 async global->LDS copy (ASYNCcnt-tracked), with synchronous fallback.
// Builtin signature (confirmed by toolchain diagnostic):
//   void __builtin_amdgcn_global_load_async_to_lds_b128(int4 AS1*, int4 AS3*,
//                                                       imm int offset, imm int cpol)
// ---------------------------------------------------------------------------
#if defined(__has_builtin)
#if __has_builtin(__builtin_amdgcn_global_load_async_to_lds_b128)
#define ASYNC_COPY 1
#else
#define ASYNC_COPY 0
#endif
#else
#define ASYNC_COPY 0
#endif

// Copy 16 bytes global -> LDS (per lane).
__device__ __forceinline__ void copy16_to_lds(const bf16* __restrict__ g,
                                              bf16* __restrict__ l) {
#if ASYNC_COPY
  __builtin_amdgcn_global_load_async_to_lds_b128((as1_v4i)(v4i*)(void*)g,
                                                 (as3_v4i)(v4i*)(void*)l, 0, 0);
#else
  *(v8bf*)l = *(const v8bf*)g;
#endif
}

__device__ __forceinline__ void async_wait0() {
#if ASYNC_COPY
#if __has_builtin(__builtin_amdgcn_s_wait_asynccnt)
  __builtin_amdgcn_s_wait_asynccnt(0);
#else
  asm volatile("s_wait_asynccnt 0x0" ::: "memory");
#endif
#endif
}

// ---------------------------------------------------------------------------
// WMMA helpers (CDNA5 wave32, 16x16x32 bf16 -> f32)
// ---------------------------------------------------------------------------
__device__ __forceinline__ v8f wmma_bf16(v16bf a, v16bf b, v8f c) {
  return __builtin_amdgcn_wmma_f32_16x16x32_bf16(false, a, false, b, (short)0, c,
                                                 false, false);
}

// Load a 16x32 bf16 A/B fragment. Per ISA layout: lanes 0-15 hold row (l&15),
// K=0..7 (regs 0-3) and K=16..23 (regs 4-7); lanes 16-31 hold K=8..15 / 24..31.
__device__ __forceinline__ v16bf load_frag(const bf16* base, int ld) {
  int lane = threadIdx.x & 31;
  const bf16* p = base + (lane & 15) * ld + ((lane >> 4) << 3);
  v8bf lo = *(const v8bf*)p;          // K = kh .. kh+7
  v8bf hi = *(const v8bf*)(p + 16);   // K = kh+16 .. kh+23
  v16bf r;
#pragma unroll
  for (int i = 0; i < 8; ++i) { r[i] = lo[i]; r[i + 8] = hi[i]; }
  return r;
}

__device__ __forceinline__ v16bf pack_frag(v8bf lo, v8bf hi) {
  v16bf r;
#pragma unroll
  for (int i = 0; i < 8; ++i) { r[i] = lo[i]; r[i + 8] = hi[i]; }
  return r;
}

// ---------------------------------------------------------------------------
// RMSNorm (fp32 in) -> bf16 out.  One block per row of D=2048.
// remap_out=1: input rows are [s,b]-major, output rows are [b,s]-major.
// ---------------------------------------------------------------------------
__global__ __launch_bounds__(256)
void rmsnorm_cast_kernel(const float* __restrict__ x, const float* __restrict__ w,
                         bf16* __restrict__ out, int remap_out) {
  __shared__ float red[8];
  __shared__ float scale_s;
  const int row = blockIdx.x;
  const float* xr = x + (size_t)row * DMODEL;
  int orow_i = row;
  if (remap_out) {  // in row = s*B + b  ->  out row = b*S + s
    orow_i = (row & 1) * S_LEN + (row >> 1);
  }
  bf16* orow = out + (size_t)orow_i * DMODEL;

  float ss = 0.f;
#pragma unroll
  for (int i = 0; i < DMODEL / 256; ++i) {
    float v = xr[threadIdx.x + i * 256];
    ss += v * v;
  }
#pragma unroll
  for (int m = 16; m >= 1; m >>= 1) ss += __shfl_xor(ss, m, 32);
  if ((threadIdx.x & 31) == 0) red[threadIdx.x >> 5] = ss;
  __syncthreads();
  if (threadIdx.x == 0) {
    float t = 0.f;
#pragma unroll
    for (int i = 0; i < 8; ++i) t += red[i];
    scale_s = rsqrtf(t / (float)DMODEL + EPSF);
  }
  __syncthreads();
  const float sc = scale_s;
#pragma unroll
  for (int i = 0; i < DMODEL / 256; ++i) {
    int c = threadIdx.x + i * 256;
    orow[c] = (bf16)(w[c] * xr[c] * sc);
  }
}

// ---------------------------------------------------------------------------
// GEMM: C[M,N] = A[M,K](bf16,row-major) @ W[N,K](fp32,row-major)^T
// Block tile 128x128x32, 256 threads (8 waves), wave tile 32x64 (2x4 WMMA).
// Double-buffered LDS; A staged with async global->LDS copies, W converted
// fp32->bf16 while staging.  One barrier per K-step.
// EPI: 0 = outB[m*N+n]  = bf16(acc + bias)                      (Q/K proj)
//      1 = outB[m*N+n]  = bf16(silu(acc))                       (w1)
//      2 = outB[m*N+n] *= acc  (bf16, in-place gate multiply)   (w3)
//      3 = outF[m*N+n]  = acc + bias + resid[remap(m)*N+n]      (o-proj)
//      4 = outB[n*M+m]  = bf16(acc + bias)  (transposed store)  (V proj)
//      5 = outF[remap(m)*N+n] = acc + resid[m*N+n]              (w2/final)
// ---------------------------------------------------------------------------
template <int EPI>
__global__ __launch_bounds__(256)
void gemm_bf16_kernel(const bf16* __restrict__ A, const float* __restrict__ Bw,
                      const float* __restrict__ bias, const float* __restrict__ resid,
                      bf16* __restrict__ outB, float* __restrict__ outF,
                      int M, int N, int K) {
  constexpr int BM = 128, BN = 128, BK = 32, LDS = BK + 8;
  __shared__ bf16 As[2][BM][LDS];
  __shared__ bf16 Bs[2][BN][LDS];

  const int tid = threadIdx.x;
  const int lane = tid & 31;
  const int wave = tid >> 5;
  const int wm = wave >> 1;   // 0..3 -> 32-row slab
  const int wn = wave & 1;    // 0..1 -> 64-col slab
  const int m0 = blockIdx.y * BM;
  const int n0 = blockIdx.x * BN;

  // Per-thread staging coordinates.
  const int ar0 = (tid + 0)   >> 2, ac0 = ((tid + 0)   & 3) << 3;  // A chunk 0
  const int ar1 = (tid + 256) >> 2, ac1 = ((tid + 256) & 3) << 3;  // A chunk 1

  v8f acc[2][4];
  const v8f zf = {0.f, 0.f, 0.f, 0.f, 0.f, 0.f, 0.f, 0.f};
#pragma unroll
  for (int i = 0; i < 2; ++i)
#pragma unroll
    for (int j = 0; j < 4; ++j) acc[i][j] = zf;

  auto stage = [&](int buf, int k0) {
    // A tile: 128x32 bf16, one 16B async copy x2 per thread
    copy16_to_lds(A + (size_t)(m0 + ar0) * K + k0 + ac0, &As[buf][ar0][ac0]);
    copy16_to_lds(A + (size_t)(m0 + ar1) * K + k0 + ac1, &As[buf][ar1][ac1]);
    // B tile: 128x32 fp32 -> bf16, one float4 x4 per thread
#pragma unroll
    for (int i = 0; i < 4; ++i) {
      int idx = tid + i * 256;
      int r = idx >> 3, c = (idx & 7) << 2;
      v4f v = *(const v4f*)(Bw + (size_t)(n0 + r) * K + k0 + c);
      Bs[buf][r][c + 0] = (bf16)v[0];
      Bs[buf][r][c + 1] = (bf16)v[1];
      Bs[buf][r][c + 2] = (bf16)v[2];
      Bs[buf][r][c + 3] = (bf16)v[3];
    }
  };

  const int kTiles = K / BK;
  stage(0, 0);
  int cur = 0;

  for (int t = 0; t < kTiles; ++t) {
    async_wait0();
    __syncthreads();
    if (t + 1 < kTiles) stage(cur ^ 1, (t + 1) * BK);
    if (t + 2 < kTiles) {
      __builtin_prefetch(A + (size_t)(m0 + ar0) * K + (t + 2) * BK + ac0, 0, 1);
      __builtin_prefetch(Bw + (size_t)(n0 + (tid >> 1)) * K + (t + 2) * BK, 0, 1);
    }

    v16bf af[2], bfv[4];
#pragma unroll
    for (int i = 0; i < 2; ++i)
      af[i] = load_frag(&As[cur][wm * 32 + i * 16][0], LDS);
#pragma unroll
    for (int j = 0; j < 4; ++j)
      bfv[j] = load_frag(&Bs[cur][wn * 64 + j * 16][0], LDS);
#pragma unroll
    for (int i = 0; i < 2; ++i)
#pragma unroll
      for (int j = 0; j < 4; ++j) acc[i][j] = wmma_bf16(af[i], bfv[j], acc[i][j]);
    cur ^= 1;
  }

  // Epilogue.  C layout: lanes 0-15 -> rows v, lanes 16-31 -> rows v+8; col = lane&15.
  const int crow = (lane >> 4) << 3;
  const int ccol = lane & 15;
#pragma unroll
  for (int i = 0; i < 2; ++i) {
#pragma unroll
    for (int j = 0; j < 4; ++j) {
      const int gcol = n0 + wn * 64 + j * 16 + ccol;
      const int grow0 = m0 + wm * 32 + i * 16 + crow;
      float bv = 0.f;
      if (EPI == 0 || EPI == 3 || EPI == 4) bv = bias ? bias[gcol] : 0.f;
      if (EPI == 4) {
        // Transposed bf16 store: 8 consecutive rows of one column -> 16B vector.
        v8bf ov;
#pragma unroll
        for (int v = 0; v < 8; ++v) ov[v] = (bf16)(acc[i][j][v] + bv);
        *(v8bf*)(outB + (size_t)gcol * M + grow0) = ov;
      } else {
#pragma unroll
        for (int v = 0; v < 8; ++v) {
          const int grow = grow0 + v;
          const size_t off = (size_t)grow * N + gcol;
          const float val = acc[i][j][v] + bv;
          if (EPI == 0) {
            outB[off] = (bf16)val;
          } else if (EPI == 1) {
            outB[off] = (bf16)(val / (1.f + __expf(-val)));
          } else if (EPI == 2) {
            outB[off] = (bf16)((float)outB[off] * val);
          } else if (EPI == 3) {
            outF[off] = val + resid[remap_row(grow) * N + gcol];
          } else {  // EPI == 5
            outF[remap_row(grow) * N + gcol] = val + resid[off];
          }
        }
      }
    }
  }
}

// ---------------------------------------------------------------------------
// Causal flash attention.  Grid: (S/128, B*H).  256 threads = 8 waves; each
// wave owns 16 query rows (128-row Q tile per block).  K and pre-transposed V
// tiles (64 keys) are double-buffered and async-staged; one barrier per tile.
// Masked (diagonal) tiles are dispatched through a compile-time-bool body so
// the main loop carries no mask compares.
// Q,K layout: [b, s, h, dh] (row = b*S+s, stride D).  Vt layout: [h, dh, b, s].
// O (ctx): [b, s, h, dh].
// ---------------------------------------------------------------------------
__global__ __launch_bounds__(256)
void attention_kernel(const bf16* __restrict__ Q, const bf16* __restrict__ Kb,
                      const bf16* __restrict__ Vt, bf16* __restrict__ O) {
  __shared__ bf16 Ks[2][64][DHEAD + 8];    // [buf][key][dh]
  __shared__ bf16 Vs[2][DHEAD][64 + 8];    // [buf][dh][key] (already transposed)
  __shared__ bf16 Ps[8][16][64 + 8];       // per-wave P scratch [qrow][key]

  const int tid = threadIdx.x;
  const int lane = tid & 31;
  const int wave = tid >> 5;
  const int bx = blockIdx.x;
  const int h = blockIdx.y % NHEAD;
  const int b = blockIdx.y / NHEAD;
  const int qs0 = bx * 128 + wave * 16;    // this wave's first query s-index
  const int crow = (lane >> 4) << 3;
  const int ccol = lane & 15;
  const float scale = 0.08838834764831845f;  // 1/sqrt(128)

  // Q fragments (16 rows x 128 dh -> 4 frags), loaded from global, pre-scaled.
  v16bf qf[4];
  {
    const int r = lane & 15, kh = (lane >> 4) << 3;
    const bf16* qp = Q + ((size_t)(b * S_LEN + qs0 + r)) * DMODEL + h * DHEAD + kh;
#pragma unroll
    for (int f = 0; f < 4; ++f) {
      qf[f] = pack_frag(*(const v8bf*)(qp + f * 32), *(const v8bf*)(qp + f * 32 + 16));
#pragma unroll
      for (int i = 0; i < 16; ++i) qf[f][i] = (bf16)((float)qf[f][i] * scale);
    }
  }

  const v8f zf = {0.f, 0.f, 0.f, 0.f, 0.f, 0.f, 0.f, 0.f};
  v8f oacc[8];
#pragma unroll
  for (int t = 0; t < 8; ++t) oacc[t] = zf;
  float mrow[8], lrow[8];
#pragma unroll
  for (int v = 0; v < 8; ++v) { mrow[v] = -__builtin_inff(); lrow[v] = 0.f; }

  auto stage = [&](int buf, int kb0) {
    // K tile: 64 keys x 128 dh, contiguous 16B chunks (async)
#pragma unroll
    for (int i = 0; i < 4; ++i) {
      int idx = tid + i * 256;
      int r = idx >> 4, c = (idx & 15) << 3;
      copy16_to_lds(Kb + ((size_t)(b * S_LEN + kb0 + r)) * DMODEL + h * DHEAD + c,
                    &Ks[buf][r][c]);
    }
    // pre-transposed V tile: 128 dh x 64 keys, contiguous 16B chunks (async)
#pragma unroll
    for (int i = 0; i < 4; ++i) {
      int idx = tid + i * 256;
      int r = idx >> 3, c = (idx & 7) << 3;
      copy16_to_lds(Vt + (size_t)(h * DHEAD + r) * MROWS + b * S_LEN + kb0 + c,
                    &Vs[buf][r][c]);
    }
  };

  auto tile_body = [&](int kb0, int buf, auto maskc) {
    constexpr bool MASK = decltype(maskc)::value;
    // S = Q @ K^T  (4 n-subtiles x 4 k-steps)
    v8f sacc[4];
#pragma unroll
    for (int t = 0; t < 4; ++t) {
      sacc[t] = zf;
#pragma unroll
      for (int f = 0; f < 4; ++f) {
        v16bf kf = load_frag(&Ks[buf][t * 16][f * 32], DHEAD + 8);
        sacc[t] = wmma_bf16(qf[f], kf, sacc[t]);
      }
    }
    if (MASK) {
#pragma unroll
      for (int t = 0; t < 4; ++t)
#pragma unroll
        for (int v = 0; v < 8; ++v)
          if ((kb0 + t * 16 + ccol) > (qs0 + crow + v)) sacc[t][v] = -__builtin_inff();
    }

    // online softmax: row max across 4 tiles and 16 lanes of the half-group
    float cand[8];
#pragma unroll
    for (int v = 0; v < 8; ++v)
      cand[v] = fmaxf(fmaxf(sacc[0][v], sacc[1][v]), fmaxf(sacc[2][v], sacc[3][v]));
#pragma unroll
    for (int m = 8; m >= 1; m >>= 1)
#pragma unroll
      for (int v = 0; v < 8; ++v) cand[v] = fmaxf(cand[v], __shfl_xor(cand[v], m, 32));

    float alpha[8];
#pragma unroll
    for (int v = 0; v < 8; ++v) {
      float mn = fmaxf(mrow[v], cand[v]);
      alpha[v] = __expf(mrow[v] - mn);
      mrow[v] = mn;
    }

    float rs[8];
#pragma unroll
    for (int v = 0; v < 8; ++v) rs[v] = 0.f;
#pragma unroll
    for (int t = 0; t < 4; ++t)
#pragma unroll
      for (int v = 0; v < 8; ++v) {
        float p = __expf(sacc[t][v] - mrow[v]);
        rs[v] += p;
        Ps[wave][crow + v][t * 16 + ccol] = (bf16)p;  // C-layout -> LDS
      }
#pragma unroll
    for (int m = 8; m >= 1; m >>= 1)
#pragma unroll
      for (int v = 0; v < 8; ++v) rs[v] += __shfl_xor(rs[v], m, 32);
#pragma unroll
    for (int v = 0; v < 8; ++v) lrow[v] = lrow[v] * alpha[v] + rs[v];

    // rescale running O
#pragma unroll
    for (int t = 0; t < 8; ++t)
#pragma unroll
      for (int v = 0; v < 8; ++v) oacc[t][v] *= alpha[v];

    // O += P @ V  (A-frags of P from LDS; B-frags from transposed V)
    v16bf pf[2];
    pf[0] = load_frag(&Ps[wave][0][0], 64 + 8);
    pf[1] = load_frag(&Ps[wave][0][32], 64 + 8);
#pragma unroll
    for (int t = 0; t < 8; ++t)
#pragma unroll
      for (int f = 0; f < 2; ++f) {
        v16bf vf = load_frag(&Vs[buf][t * 16][f * 32], 64 + 8);
        oacc[t] = wmma_bf16(pf[f], vf, oacc[t]);
      }
  };

  const int nTiles = 2 * bx + 2;   // key tiles covering [0, bx*128+127]
  stage(0, 0);
  int cur = 0;
  for (int j = 0; j < nTiles; ++j) {
    async_wait0();
    __syncthreads();
    if (j + 1 < nTiles) stage(cur ^ 1, (j + 1) * 64);
    if (j + 2 < nTiles) {                 // last two tiles touch the diagonal
      tile_body(j * 64, cur, BoolC<false>{});
    } else {
      tile_body(j * 64, cur, BoolC<true>{});
    }
    cur ^= 1;
  }

  // normalize and write ctx (bf16, [b,s] rows)
#pragma unroll
  for (int t = 0; t < 8; ++t)
#pragma unroll
    for (int v = 0; v < 8; ++v) {
      float o = oacc[t][v] / lrow[v];
      O[((size_t)(b * S_LEN + qs0 + crow + v)) * DMODEL + h * DHEAD + t * 16 + ccol] =
          (bf16)o;
    }
}

// ---------------------------------------------------------------------------
// Host-side orchestration
// ---------------------------------------------------------------------------
extern "C" void kernel_launch(void* const* d_in, const int* in_sizes, int n_in,
                              void* d_out, int out_size, void* d_ws, size_t ws_size,
                              hipStream_t stream) {
  const float* x       = (const float*)d_in[0];
  const float* norm1_w = (const float*)d_in[1];
  const float* norm2_w = (const float*)d_in[2];
  const float* q_w = (const float*)d_in[3];
  const float* q_b = (const float*)d_in[4];
  const float* k_w = (const float*)d_in[5];
  const float* k_b = (const float*)d_in[6];
  const float* v_w = (const float*)d_in[7];
  const float* v_b = (const float*)d_in[8];
  const float* o_w = (const float*)d_in[9];
  const float* o_b = (const float*)d_in[10];
  const float* w1  = (const float*)d_in[11];
  const float* w2  = (const float*)d_in[12];
  const float* w3  = (const float*)d_in[13];
  float* out = (float*)d_out;

  char* ws = (char*)d_ws;
  size_t off = 0;
  auto wsalloc = [&](size_t bytes) -> char* {
    char* p = ws + off;
    off += (bytes + 255) & ~(size_t)255;
    return p;
  };
  bf16* h1  = (bf16*)wsalloc((size_t)MROWS * DMODEL * 2);  // [b,s] rows
  bf16* qb  = (bf16*)wsalloc((size_t)MROWS * DMODEL * 2);  // [b,s] rows
  bf16* kb  = (bf16*)wsalloc((size_t)MROWS * DMODEL * 2);  // [b,s] rows
  bf16* vtb = (bf16*)wsalloc((size_t)MROWS * DMODEL * 2);  // transposed [n][m]
  bf16* ctx = (bf16*)wsalloc((size_t)MROWS * DMODEL * 2);  // [b,s] rows
  bf16* h2  = (bf16*)wsalloc((size_t)MROWS * DMODEL * 2);  // [b,s] rows
  float* x2 = (float*)wsalloc((size_t)MROWS * DMODEL * 4); // [b,s] rows
  bf16* g   = (bf16*)wsalloc((size_t)MROWS * HIDDEN * 2);  // [b,s] rows

  const dim3 blk256(256);
  const dim3 gD(DMODEL / 128, MROWS / 128);     // N=2048 GEMMs
  const dim3 gH(HIDDEN / 128, MROWS / 128);     // N=5632 GEMMs
  const dim3 gAtt(S_LEN / 128, BATCH * NHEAD);

  // 1) h1 = bf16(rmsnorm(x, norm1_w)), remapped [s,b] -> [b,s]
  rmsnorm_cast_kernel<<<MROWS, blk256, 0, stream>>>(x, norm1_w, h1, 1);
  // 2) q/k = h1 @ W^T + b (bf16);  v stored transposed [n][m]
  gemm_bf16_kernel<0><<<gD, blk256, 0, stream>>>(h1, q_w, q_b, nullptr, qb, nullptr,
                                                 MROWS, DMODEL, DMODEL);
  gemm_bf16_kernel<0><<<gD, blk256, 0, stream>>>(h1, k_w, k_b, nullptr, kb, nullptr,
                                                 MROWS, DMODEL, DMODEL);
  gemm_bf16_kernel<4><<<gD, blk256, 0, stream>>>(h1, v_w, v_b, nullptr, vtb, nullptr,
                                                 MROWS, DMODEL, DMODEL);
  // 3) ctx = causal_flash_attention(q, k, v^T)
  attention_kernel<<<gAtt, blk256, 0, stream>>>(qb, kb, vtb, ctx);
  // 4) x2 = x(remapped) + ctx @ o_w^T + o_b   (x2 kept in [b,s] layout)
  gemm_bf16_kernel<3><<<gD, blk256, 0, stream>>>(ctx, o_w, o_b, x, nullptr, x2,
                                                 MROWS, DMODEL, DMODEL);
  // 5) h2 = bf16(rmsnorm(x2, norm2_w))   ([b,s] in, [b,s] out)
  rmsnorm_cast_kernel<<<MROWS, blk256, 0, stream>>>(x2, norm2_w, h2, 0);
  // 6) g = silu(h2 @ w1^T)
  gemm_bf16_kernel<1><<<gH, blk256, 0, stream>>>(h2, w1, nullptr, nullptr, g, nullptr,
                                                 MROWS, HIDDEN, DMODEL);
  // 7) g *= h2 @ w3^T   (in place)
  gemm_bf16_kernel<2><<<gH, blk256, 0, stream>>>(h2, w3, nullptr, nullptr, g, nullptr,
                                                 MROWS, HIDDEN, DMODEL);
  // 8) out[s,b] = x2 + g @ w2^T   (store remapped back to [S,B,D])
  gemm_bf16_kernel<5><<<gD, blk256, 0, stream>>>(g, w2, nullptr, x2, nullptr, out,
                                                 MROWS, DMODEL, HIDDEN);
  (void)in_sizes; (void)n_in; (void)out_size; (void)ws_size;
}
